// OptimizedMoE2_22222160789643
// MI455X (gfx1250) — matrix operand order repaired and checked by reading, not compiled, not run on hardware
//
#include <hip/hip_runtime.h>

#define N_TOK   16384
#define D_INF   1024
#define D_OUTF  1024
#define HDIM    2048
#define NEXP    16
#define NSLOTS  (N_TOK*2)

typedef __attribute__((ext_vector_type(16))) __bf16 v16bf;
typedef __attribute__((ext_vector_type(8)))  float  v8f;
typedef unsigned short u16;
typedef unsigned int   u32;
typedef __attribute__((ext_vector_type(8)))  u16 v8us;
typedef __attribute__((ext_vector_type(16))) u16 v16us;

__device__ inline u16 f2bf(float f) {
  u32 u = __builtin_bit_cast(u32, f);
  u += 0x7fffu + ((u >> 16) & 1u);   // round-to-nearest-even
  return (u16)(u >> 16);
}

__device__ inline v16bf bfcomb(v8us lo, v8us hi) {
  v16us c;
#pragma unroll
  for (int i = 0; i < 8; i++) { c[i] = lo[i]; c[i + 8] = hi[i]; }
  return __builtin_bit_cast(v16bf, c);
}

// CDNA5 async copy: 16B global -> LDS, tracked by ASYNCcnt (no VGPR round-trip)
__device__ inline void async_g2l_b128(unsigned lds_off, const void* gaddr) {
  asm volatile("global_load_async_to_lds_b128 %0, %1, off"
               :: "v"(lds_off), "v"(gaddr) : "memory");
}
__device__ inline void wait_async0() {
  asm volatile("s_wait_asynccnt 0x0" ::: "memory");
}

// ---------------------------------------------------------------- prep: x -> bf16, zero counters
__global__ __launch_bounds__(256)
void prep_kernel(const float* __restrict__ x, u16* __restrict__ xb, int* __restrict__ cnt) {
  long idx = (long)blockIdx.x * 256 + threadIdx.x;
  if (idx < (long)N_TOK * D_INF) xb[idx] = f2bf(x[idx]);
  if (blockIdx.x == 0 && threadIdx.x < NEXP) cnt[threadIdx.x] = 0;
}

// ---------------------------------------------------------------- transpose+convert W[K][N] f32 -> WT[N][K] bf16
__global__ __launch_bounds__(256)
void transw_kernel(const float* __restrict__ W, u16* __restrict__ WT, int K, int Nw) {
  __shared__ float tile[32][33];
  const int e = blockIdx.z;
  const int n0 = blockIdx.x * 32;
  const int k0 = blockIdx.y * 32;
  const float* We = W + (size_t)e * K * Nw;
  u16* WTe = WT + (size_t)e * Nw * K;
  const int t = threadIdx.x;
#pragma unroll
  for (int i = 0; i < 4; i++) {
    int idx = i * 256 + t;
    int r = idx >> 5, c = idx & 31;                      // r: k, c: n (coalesced read)
    tile[r][c] = We[(size_t)(k0 + r) * Nw + n0 + c];
  }
  __syncthreads();
#pragma unroll
  for (int i = 0; i < 4; i++) {
    int idx = i * 256 + t;
    int r = idx >> 5, c = idx & 31;                      // r: n, c: k (coalesced write)
    WTe[(size_t)(n0 + r) * K + k0 + c] = f2bf(tile[c][r]);
  }
}

// ---------------------------------------------------------------- gating: logits, top-2, softmax
__global__ __launch_bounds__(256)
void gating_kernel(const float* __restrict__ x, const float* __restrict__ Wg,
                   const float* __restrict__ bg, int* __restrict__ cnt,
                   int* __restrict__ topE, float* __restrict__ topG) {
  const int lane = threadIdx.x & 31;
  const int n = blockIdx.x * 8 + (threadIdx.x >> 5);
  if (n >= N_TOK) return;
  float acc[NEXP];
#pragma unroll
  for (int e = 0; e < NEXP; e++) acc[e] = 0.f;
  const float* xr = x + (long)n * D_INF;
  for (int d = lane; d < D_INF; d += 32) {
    const float xv = xr[d];
    const float* wr = Wg + (long)d * NEXP;
#pragma unroll
    for (int e = 0; e < NEXP; e++) acc[e] += xv * wr[e];
  }
#pragma unroll
  for (int off = 16; off >= 1; off >>= 1) {
#pragma unroll
    for (int e = 0; e < NEXP; e++) acc[e] += __shfl_xor(acc[e], off, 32);
  }
  if (lane == 0) {
    float best = -3.4e38f, sec = -3.4e38f; int be = 0, se = 0;
#pragma unroll
    for (int e = 0; e < NEXP; e++) {
      float l = acc[e] + bg[e];
      if (l > best)     { sec = best; se = be; best = l; be = e; }
      else if (l > sec) { sec = l; se = e; }
    }
    float e1 = expf(sec - best);
    float inv = 1.f / (1.f + e1);
    atomicAdd(&cnt[be], 1);
    atomicAdd(&cnt[se], 1);
    topE[2 * n] = be;  topE[2 * n + 1] = se;
    topG[2 * n] = inv; topG[2 * n + 1] = e1 * inv;
  }
}

// ---------------------------------------------------------------- exclusive scan of 16 counts
__global__ void scan_kernel(const int* __restrict__ cnt, int* __restrict__ offs,
                            int* __restrict__ fill) {
  if (threadIdx.x == 0) {
    int a = 0;
    for (int e = 0; e < NEXP; e++) { offs[e] = a; a += cnt[e]; }
  }
  if (threadIdx.x < NEXP) fill[threadIdx.x] = 0;
}

// ---------------------------------------------------------------- build per-expert token lists
__global__ __launch_bounds__(256)
void scatter_kernel(const int* __restrict__ topE, const int* __restrict__ offs,
                    int* __restrict__ fill, int* __restrict__ tokens,
                    int* __restrict__ slotOf) {
  int n = blockIdx.x * 256 + threadIdx.x;
  if (n >= N_TOK) return;
#pragma unroll
  for (int c = 0; c < 2; c++) {
    int e = topE[2 * n + c];
    int s = offs[e] + atomicAdd(&fill[e], 1);
    tokens[s] = n;
    slotOf[2 * n + c] = s;
  }
}

// ---------------------------------------------------------------- WMMA GEMM (double-buffered async-LDS)
// MODE 0: A = gather(xb via token list), relu -> bf16 out
// MODE 1: A = slot rows,                relu -> bf16 out
// MODE 2: A = slot rows,                no relu -> f32 out
// BT is pre-transposed bf16 weights: [E][N][K] (per-column contiguous K)
#define KTILE 64
#define LSTR  72   // u16/row: 144B pitch -> 16B-aligned chunks, bank-stride 36 (conflict-free)
#define TILE_E (128 * LSTR)          // u16 elements per tile buffer
#define TILE_B (TILE_E * 2)          // bytes per tile buffer

template <int MODE>
__global__ __launch_bounds__(256)
void moe_gemm(const u16* __restrict__ A, int lda,
              const u16* __restrict__ BT, int Nw,
              const float* __restrict__ bias, int biasStrideE,
              u16* __restrict__ Yb, float* __restrict__ Yf, int ldy, int K,
              const int* __restrict__ cnt, const int* __restrict__ offs,
              const int* __restrict__ tokens) {
  __shared__ u16 lds_a[2 * TILE_E];
  __shared__ u16 lds_b[2 * TILE_E];

  const int e = blockIdx.z;
  const int cntE = cnt[e];
  const int m0 = blockIdx.y * 128;
  if (m0 >= cntE) return;                 // uniform per block
  const int offsE = offs[e];
  const int n0 = blockIdx.x * 128;

  const int tid  = threadIdx.x;
  const int lane = tid & 31;
  const int wid  = tid >> 5;
  const int half = lane >> 4;             // K-half select per WMMA layout
  const int l16  = lane & 15;
  const int wm   = wid >> 1;              // wave M base = wm*32
  const int wn   = wid & 1;               // wave N base = wn*64

  const u16* BTe = BT + (size_t)e * Nw * K;

  // Async-chunk assignment: tile = 128 rows x 64 u16 = 8 chunks/row of 16B.
  // Thread (crow, cpart) covers rows {c*32+crow}, chunk cpart; 4 chunks/tile/thread.
  const int crow  = tid >> 3;             // 0..31
  const int cpart = tid & 7;              // 0..7
  const u16* agp[4];
  const u16* bgp[4];
  unsigned aldo[4], bldo[4];              // LDS byte offsets within buffer 0
#pragma unroll
  for (int c = 0; c < 4; c++) {
    int r = c * 32 + crow;
    int mrow = m0 + r;
    int mr = mrow < cntE ? mrow : cntE - 1;   // clamp: never read invalid rows
    long arow = (MODE == 0) ? (long)tokens[offsE + mr] : (long)(offsE + mr);
    agp[c] = A + arow * (size_t)lda + cpart * 8;
    bgp[c] = BTe + (size_t)(n0 + r) * K + cpart * 8;
    aldo[c] = (unsigned)(size_t)(void*)&lds_a[r * LSTR + cpart * 8];
    bldo[c] = (unsigned)(size_t)(void*)&lds_b[r * LSTR + cpart * 8];
  }

  const v8f vz = {0.f, 0.f, 0.f, 0.f, 0.f, 0.f, 0.f, 0.f};
  v8f acc[2][4];
#pragma unroll
  for (int i = 0; i < 2; i++)
#pragma unroll
    for (int j = 0; j < 4; j++) acc[i][j] = vz;

  // Prologue: stage tile 0 into buffer 0
#pragma unroll
  for (int c = 0; c < 4; c++) {
    async_g2l_b128(aldo[c], agp[c]);
    async_g2l_b128(bldo[c], bgp[c]);
    agp[c] += KTILE; bgp[c] += KTILE;
  }

  for (int k0 = 0; k0 < K; k0 += KTILE) {
    const unsigned curB = ((k0 / KTILE) & 1) ? (unsigned)TILE_B : 0u;
    wait_async0();          // tile k0 landed in LDS (in-order async completion)
    __syncthreads();        // published to all waves

    if (k0 + KTILE < K) {   // prefetch tile k0+1 into the other buffer (overlaps WMMAs)
      const unsigned nxtB = curB ^ (unsigned)TILE_B;
#pragma unroll
      for (int c = 0; c < 4; c++) {
        async_g2l_b128(aldo[c] + nxtB, agp[c]);
        async_g2l_b128(bldo[c] + nxtB, bgp[c]);
        agp[c] += KTILE; bgp[c] += KTILE;
      }
    }

    const u16* La = lds_a + (curB >> 1);
    const u16* Lb = lds_b + (curB >> 1);
#pragma unroll
    for (int s = 0; s < 2; s++) {          // two K=32 WMMA steps per LDS tile
      v16bf af[2], bfv[4];
#pragma unroll
      for (int i = 0; i < 2; i++) {
        const u16* rp = La + (wm * 32 + i * 16 + l16) * LSTR + s * 32;
        af[i] = bfcomb(*(const v8us*)(rp + half * 8), *(const v8us*)(rp + 16 + half * 8));
      }
#pragma unroll
      for (int j = 0; j < 4; j++) {
        const u16* cp = Lb + (wn * 64 + j * 16 + l16) * LSTR + s * 32;
        bfv[j] = bfcomb(*(const v8us*)(cp + half * 8), *(const v8us*)(cp + 16 + half * 8));
      }
#pragma unroll
      for (int i = 0; i < 2; i++)
#pragma unroll
        for (int j = 0; j < 4; j++)
          acc[i][j] = __builtin_amdgcn_wmma_f32_16x16x32_bf16(
              false, af[i], false, bfv[j], (short)0, acc[i][j], false, false);
    }
    __syncthreads();   // wmma's dscnt waits ensure frag reads done before buffer reuse
  }

  const float* biasE = bias + (long)e * biasStrideE;
  // Epilogue: C layout -> VGPR v holds row (v + half*8), col = l16
#define MOE_EPI(GUARD)                                                        \
  _Pragma("unroll")                                                           \
  for (int i = 0; i < 2; i++) {                                               \
    _Pragma("unroll")                                                         \
    for (int j = 0; j < 4; j++) {                                             \
      const int ncol = n0 + wn * 64 + j * 16 + l16;                           \
      _Pragma("unroll")                                                       \
      for (int v = 0; v < 8; v++) {                                           \
        int mrow = m0 + wm * 32 + i * 16 + half * 8 + v;                      \
        if (!GUARD || mrow < cntE) {                                          \
          float y = acc[i][j][v] + biasE[ncol];                               \
          if (MODE < 2) {                                                     \
            y = fmaxf(y, 0.f);                                                \
            Yb[(long)(offsE + mrow) * ldy + ncol] = f2bf(y);                  \
          } else {                                                            \
            Yf[(long)(offsE + mrow) * ldy + ncol] = y;                        \
          }                                                                   \
        }                                                                     \
      }                                                                       \
    }                                                                         \
  }
  if (m0 + 128 <= cntE) { MOE_EPI(false) }   // full tile: straight-line stores
  else                  { MOE_EPI(true)  }   // ragged tail: guarded stores
#undef MOE_EPI
}

// ---------------------------------------------------------------- combine: deterministic, no atomics
__global__ __launch_bounds__(256)
void combine_kernel(const float* __restrict__ Y3, const int* __restrict__ slotOf,
                    const float* __restrict__ topG, float* __restrict__ out) {
  long idx = (long)blockIdx.x * 256 + threadIdx.x;
  if (idx >= (long)N_TOK * D_OUTF) return;
  int n = (int)(idx >> 10);     // D_OUTF == 1024
  int d = (int)(idx & 1023);
  int s0 = slotOf[2 * n], s1 = slotOf[2 * n + 1];
  out[idx] = topG[2 * n] * Y3[(long)s0 * D_OUTF + d] +
             topG[2 * n + 1] * Y3[(long)s1 * D_OUTF + d];
}

extern "C" void kernel_launch(void* const* d_in, const int* in_sizes, int n_in,
                              void* d_out, int out_size, void* d_ws, size_t ws_size,
                              hipStream_t stream) {
  const float* x  = (const float*)d_in[0];
  const float* Wg = (const float*)d_in[1];
  const float* bg = (const float*)d_in[2];
  const float* W1 = (const float*)d_in[3];
  const float* b1 = (const float*)d_in[4];
  const float* W2 = (const float*)d_in[5];
  const float* b2 = (const float*)d_in[6];
  const float* W3 = (const float*)d_in[7];
  const float* b3 = (const float*)d_in[8];
  float* out = (float*)d_out;

  char* p = (char*)d_ws;
  int*   cnt    = (int*)p;   p += 256;
  int*   offs   = (int*)p;   p += 256;
  int*   fill   = (int*)p;   p += 256;
  int*   tokens = (int*)p;   p += (size_t)NSLOTS * sizeof(int);
  int*   slotOf = (int*)p;   p += (size_t)NSLOTS * sizeof(int);
  int*   topE   = (int*)p;   p += (size_t)NSLOTS * sizeof(int);
  float* topG   = (float*)p; p += (size_t)NSLOTS * sizeof(float);
  u16*   xb     = (u16*)p;   p += (size_t)N_TOK * D_INF * sizeof(u16);
  u16*   w1t    = (u16*)p;   p += (size_t)NEXP * D_INF * HDIM * sizeof(u16);
  u16*   w2t    = (u16*)p;   p += (size_t)NEXP * HDIM * HDIM * sizeof(u16);
  u16*   w3t    = (u16*)p;   p += (size_t)NEXP * HDIM * D_OUTF * sizeof(u16);
  u16*   h1     = (u16*)p;   p += (size_t)NSLOTS * HDIM * sizeof(u16);
  u16*   h2     = (u16*)p;   p += (size_t)NSLOTS * HDIM * sizeof(u16);
  float* Y3     = (float*)h1;   // reuse h1 (both 128 MB); h1 dead after layer 2

  prep_kernel<<<(N_TOK * D_INF + 255) / 256, 256, 0, stream>>>(x, xb, cnt);
  transw_kernel<<<dim3(HDIM / 32, D_INF / 32, NEXP), 256, 0, stream>>>(W1, w1t, D_INF, HDIM);
  transw_kernel<<<dim3(HDIM / 32, HDIM / 32, NEXP), 256, 0, stream>>>(W2, w2t, HDIM, HDIM);
  transw_kernel<<<dim3(D_OUTF / 32, HDIM / 32, NEXP), 256, 0, stream>>>(W3, w3t, HDIM, D_OUTF);

  gating_kernel<<<N_TOK / 8, 256, 0, stream>>>(x, Wg, bg, cnt, topE, topG);
  scan_kernel<<<1, 32, 0, stream>>>(cnt, offs, fill);
  scatter_kernel<<<(N_TOK + 255) / 256, 256, 0, stream>>>(topE, offs, fill, tokens, slotOf);

  dim3 g12(HDIM / 128, N_TOK / 128, NEXP);
  moe_gemm<0><<<g12, 256, 0, stream>>>(xb, D_INF, w1t, HDIM, b1, HDIM,
                                       h1, nullptr, HDIM, D_INF, cnt, offs, tokens);
  moe_gemm<1><<<g12, 256, 0, stream>>>(h1, HDIM, w2t, HDIM, b2, HDIM,
                                       h2, nullptr, HDIM, HDIM, cnt, offs, tokens);
  dim3 g3(D_OUTF / 128, N_TOK / 128, NEXP);
  moe_gemm<2><<<g3, 256, 0, stream>>>(h2, HDIM, w3t, D_OUTF, b3, D_OUTF,
                                      nullptr, Y3, D_OUTF, HDIM, cnt, offs, tokens);
  combine_kernel<<<(N_TOK * D_OUTF + 255) / 256, 256, 0, stream>>>(Y3, slotOf, topG, out);
}
